// RNN_decoder_87737591923154
// MI455X (gfx1250) — matrix operand order repaired and checked by reading, not compiled, not run on hardware
//
#include <hip/hip_runtime.h>

// 2-layer tanh RNN, T=4096, B=32, D=H=256 (torch.nn.RNN semantics).
// Two pipelined workgroups (one WGP each), 256 threads = 8 wave32 each:
//   block 0: layer 0 -> writes h0(t) bf16 to a global ring (d_ws)
//   block 1: layer 1 -> consumes ring, writes f32 output
// Per-layer weights (W_ih,W_hh) live in LDS as bf16 WMMA B-fragments in
// fragment-major order (conflict-free ds_load_b128, no VGPR spills).
// Each wave owns TWO 16-col N-strips => every A fragment feeds 2 WMMAs.

#define TT 4096
#define BB 32
#define DD 256
#define HH 256
#define RS 264            // padded activation row stride (bf16 elems)
#define RING 4
#define SLOT (BB * HH)    // bf16 elems per ring slot

typedef __attribute__((ext_vector_type(16))) __bf16 v16bf;
typedef __attribute__((ext_vector_type(8)))  __bf16 v8bf;
typedef __attribute__((ext_vector_type(8)))  float  v8f;

__device__ __forceinline__ v8f wmma_bf16(v16bf a, v16bf b, v8f c) {
  return __builtin_amdgcn_wmma_f32_16x16x32_bf16(false, a, false, b,
                                                 (short)0, c, false, false);
}

__device__ __forceinline__ v16bf cat16(v8bf lo, v8bf hi) {
  return __builtin_shufflevector(lo, hi, 0, 1, 2, 3, 4, 5, 6, 7,
                                 8, 9, 10, 11, 12, 13, 14, 15);
}

// A-fragment (16x32 bf16) from padded LDS activation matrix.
__device__ __forceinline__ v16bf loadA_lds(const __bf16* base, int m0, int k0,
                                           int r, int half) {
  const __bf16* p = base + (m0 + r) * RS + k0 + half * 8;
  return cat16(*(const v8bf*)(p), *(const v8bf*)(p + 16));
}

// B-fragment from fragment-major packed LDS weights: per (mat,strip,chunk)
// 512 bf16 = [piece0: 32 lanes x 8][piece1: 32 lanes x 8]; lane*16B addressing.
__device__ __forceinline__ v16bf loadB_lds(const __bf16* Wl, int mat, int strip,
                                           int c, int lane) {
  const __bf16* p = Wl + ((((mat * 16 + strip) * 8 + c) << 9) + lane * 8);
  return cat16(*(const v8bf*)(p), *(const v8bf*)(p + 256));
}

__device__ __forceinline__ float tanh_fast(float x) {
  float e = __builtin_amdgcn_exp2f(x * 2.8853900817779268f);  // 2*log2(e)
  return 1.0f - 2.0f * __builtin_amdgcn_rcpf(e + 1.0f);
}

__device__ __forceinline__ unsigned ld_acq(const unsigned* p) {
  return __hip_atomic_load(p, __ATOMIC_ACQUIRE, __HIP_MEMORY_SCOPE_AGENT);
}
__device__ __forceinline__ void st_rel(unsigned* p, unsigned v) {
  __hip_atomic_store(p, v, __ATOMIC_RELEASE, __HIP_MEMORY_SCOPE_AGENT);
}

// Pack this wave's two strips of a [256,256] f32 weight matrix into
// fragment-major bf16 LDS. (Y = X @ W^T: lane n=strip*16+r holds 16
// K-values of row n; lanes 0-15 K=k0..+7,k0+16..23, lanes 16-31 K=k0+8..15,k0+24..31.)
__device__ __forceinline__ void pack_weights(const float* __restrict__ W,
                                             __bf16* Wl, int mat, int w,
                                             int lane, int r, int half) {
#pragma unroll
  for (int si = 0; si < 2; ++si) {
    const int strip = 2 * w + si;
#pragma unroll
    for (int c = 0; c < 8; ++c) {
      const float* p = W + (strip * 16 + r) * 256 + c * 32 + half * 8;
      v8bf lo = {(__bf16)p[0], (__bf16)p[1], (__bf16)p[2], (__bf16)p[3],
                 (__bf16)p[4], (__bf16)p[5], (__bf16)p[6], (__bf16)p[7]};
      v8bf hi = {(__bf16)p[16], (__bf16)p[17], (__bf16)p[18], (__bf16)p[19],
                 (__bf16)p[20], (__bf16)p[21], (__bf16)p[22], (__bf16)p[23]};
      __bf16* dst = Wl + ((((mat * 16 + strip) * 8 + c) << 9) + lane * 8);
      *(v8bf*)dst = lo;
      *(v8bf*)(dst + 256) = hi;
    }
  }
}

// Stage 4 rows (wave w -> rows 4w..4w+3) of a [32,256] f32 matrix into padded LDS.
__device__ __forceinline__ void stage4_f32(const float* __restrict__ src,
                                           __bf16* dst, int w, int lane) {
#pragma unroll
  for (int j = 0; j < 4; ++j) {
    const int row = 4 * w + j;
    const float4* p = (const float4*)(src + row * 256 + lane * 8);
    float4 f0 = p[0], f1 = p[1];
    v8bf b = {(__bf16)f0.x, (__bf16)f0.y, (__bf16)f0.z, (__bf16)f0.w,
              (__bf16)f1.x, (__bf16)f1.y, (__bf16)f1.z, (__bf16)f1.w};
    *(v8bf*)(dst + row * RS + lane * 8) = b;
  }
}

__global__ void flags_init(unsigned* f) {
  if (threadIdx.x < 2) f[threadIdx.x * 32] = 0u;  // prod, cons
}

__global__ __launch_bounds__(256, 1)
void rnn2_pipe_kernel(const float* __restrict__ x,
                      const float* __restrict__ W_ih,
                      const float* __restrict__ W_hh,
                      const float* __restrict__ b_ih,
                      const float* __restrict__ b_hh,
                      const float* __restrict__ h_init,
                      float* __restrict__ out,
                      unsigned* __restrict__ prod,
                      unsigned* __restrict__ cons,
                      __bf16* __restrict__ ring) {
  __shared__ __align__(16) __bf16 Wlds[131072];      // 256 KB packed weights
  __shared__ __align__(16) __bf16 inbuf[BB * RS];    // layer input (x or h0)
  __shared__ __align__(16) __bf16 hrec[BB * RS];     // recurrent hidden

  const int role = blockIdx.x;  // 0 = layer 0 (producer), 1 = layer 1 (consumer)
  const int tid = threadIdx.x;
  const int w = tid >> 5, lane = tid & 31;
  const int r = lane & 15, half = lane >> 4;
  const int s0 = 2 * w, s1 = 2 * w + 1;        // this wave's two N-strips
  const int col0 = s0 * 16 + r, col1 = col0 + 16;

  // ---- Prologue: pack weights, stage h(-1), stage first input ----
  pack_weights(W_ih + role * (HH * DD), Wlds, 0, w, lane, r, half);
  pack_weights(W_hh + role * (HH * HH), Wlds, 1, w, lane, r, half);
  stage4_f32(h_init + role * (BB * HH), hrec, w, lane);

  const float bias0 = b_ih[role * HH + col0] + b_hh[role * HH + col0];
  const float bias1 = b_ih[role * HH + col1] + b_hh[role * HH + col1];

  if (role == 0) {
    stage4_f32(x, inbuf, w, lane);
  } else {
    while ((int)ld_acq(prod) < 1) __builtin_amdgcn_s_sleep(2);
#pragma unroll
    for (int j = 0; j < 4; ++j) {
      const int row = 4 * w + j;
      v8bf b = *(const v8bf*)(ring + row * HH + lane * 8);
      *(v8bf*)(inbuf + row * RS + lane * 8) = b;
    }
  }
  __syncthreads();
  if (role == 1 && tid == 0) st_rel(cons, 1u);  // slot 0 consumed into LDS

  // ---- Main loop ----
  for (int t = 0; t < TT; ++t) {
    // Prefetch next input into registers (hide global latency under WMMAs).
    float4 xf[8];
    v8bf gf[4];
    const bool pre = (t + 1 < TT);
    if (role == 0) {
      // back-pressure: slot t must be free (previous occupant consumed)
      while ((int)t - (int)ld_acq(cons) >= RING) __builtin_amdgcn_s_sleep(2);
      if (pre) {
        const float* xs = x + (size_t)(t + 1) * (BB * DD);
#pragma unroll
        for (int j = 0; j < 4; ++j) {
          const float4* p = (const float4*)(xs + (4 * w + j) * 256 + lane * 8);
          xf[2 * j] = p[0];
          xf[2 * j + 1] = p[1];
        }
      }
    } else {
      if (pre) {
        while ((int)ld_acq(prod) < t + 2) __builtin_amdgcn_s_sleep(2);
        const __bf16* rs = ring + ((t + 1) & (RING - 1)) * SLOT;
#pragma unroll
        for (int j = 0; j < 4; ++j)
          gf[j] = *(const v8bf*)(rs + (4 * w + j) * HH + lane * 8);
      }
    }

    // ---- GEMMs: acc[mtile][strip] over K=256 in 8 chunks of 32 ----
    v8f a00, a01, a10, a11;
#pragma unroll
    for (int i = 0; i < 8; ++i) {
      a00[i] = bias0; a10[i] = bias0;
      a01[i] = bias1; a11[i] = bias1;
    }
#pragma unroll
    for (int c = 0; c < 8; ++c) {
      const int k0 = c * 32;
      v16bf ax0 = loadA_lds(inbuf, 0, k0, r, half);
      v16bf ax1 = loadA_lds(inbuf, 16, k0, r, half);
      v16bf bi0 = loadB_lds(Wlds, 0, s0, c, lane);
      v16bf bi1 = loadB_lds(Wlds, 0, s1, c, lane);
      a00 = wmma_bf16(ax0, bi0, a00);
      a10 = wmma_bf16(ax1, bi0, a10);
      a01 = wmma_bf16(ax0, bi1, a01);
      a11 = wmma_bf16(ax1, bi1, a11);
      v16bf ah0 = loadA_lds(hrec, 0, k0, r, half);
      v16bf ah1 = loadA_lds(hrec, 16, k0, r, half);
      v16bf bh0 = loadB_lds(Wlds, 1, s0, c, lane);
      v16bf bh1 = loadB_lds(Wlds, 1, s1, c, lane);
      a00 = wmma_bf16(ah0, bh0, a00);
      a10 = wmma_bf16(ah1, bh0, a10);
      a01 = wmma_bf16(ah0, bh1, a01);
      a11 = wmma_bf16(ah1, bh1, a11);
    }
#pragma unroll
    for (int i = 0; i < 8; ++i) {
      a00[i] = tanh_fast(a00[i]); a10[i] = tanh_fast(a10[i]);
      a01[i] = tanh_fast(a01[i]); a11[i] = tanh_fast(a11[i]);
    }

    __syncthreads();  // all reads of hrec/inbuf done

    // ---- Write phase ----
    __bf16* slot = ring + (t & (RING - 1)) * SLOT;
    float* ot = out + (size_t)t * (BB * HH);
#pragma unroll
    for (int i = 0; i < 8; ++i) {
      const int row0 = half * 8 + i, row1 = 16 + half * 8 + i;
      hrec[row0 * RS + col0] = (__bf16)a00[i];
      hrec[row1 * RS + col0] = (__bf16)a10[i];
      hrec[row0 * RS + col1] = (__bf16)a01[i];
      hrec[row1 * RS + col1] = (__bf16)a11[i];
      if (role == 0) {  // export h0(t) bf16 to ring
        slot[row0 * HH + col0] = (__bf16)a00[i];
        slot[row1 * HH + col0] = (__bf16)a10[i];
        slot[row0 * HH + col1] = (__bf16)a01[i];
        slot[row1 * HH + col1] = (__bf16)a11[i];
      } else {          // stream f32 output
        ot[row0 * HH + col0] = a00[i];
        ot[row1 * HH + col0] = a10[i];
        ot[row0 * HH + col1] = a01[i];
        ot[row1 * HH + col1] = a11[i];
      }
    }
    // Stage prefetched next input into inbuf (slot unused until next step).
    if (pre) {
      if (role == 0) {
#pragma unroll
        for (int j = 0; j < 4; ++j) {
          float4 f0 = xf[2 * j], f1 = xf[2 * j + 1];
          v8bf b = {(__bf16)f0.x, (__bf16)f0.y, (__bf16)f0.z, (__bf16)f0.w,
                    (__bf16)f1.x, (__bf16)f1.y, (__bf16)f1.z, (__bf16)f1.w};
          *(v8bf*)(inbuf + (4 * w + j) * RS + lane * 8) = b;
        }
      } else {
#pragma unroll
        for (int j = 0; j < 4; ++j)
          *(v8bf*)(inbuf + (4 * w + j) * RS + lane * 8) = gf[j];
      }
    }
    if (role == 0) __threadfence();  // make ring stores agent-visible
    __syncthreads();
    if (tid == 0) {
      if (role == 0) st_rel(prod, (unsigned)(t + 1));
      else if (pre)  st_rel(cons, (unsigned)(t + 2));  // slot t+1 consumed
    }
  }
}

extern "C" void kernel_launch(void* const* d_in, const int* in_sizes, int n_in,
                              void* d_out, int out_size, void* d_ws, size_t ws_size,
                              hipStream_t stream) {
  const float* x = (const float*)d_in[0];
  const float* W_ih = (const float*)d_in[1];
  const float* W_hh = (const float*)d_in[2];
  const float* b_ih = (const float*)d_in[3];
  const float* b_hh = (const float*)d_in[4];
  const float* h0 = (const float*)d_in[5];
  float* out = (float*)d_out;
  (void)in_sizes; (void)n_in; (void)out_size; (void)ws_size;

  unsigned* prod = (unsigned*)d_ws;                 // [0]
  unsigned* cons = (unsigned*)d_ws + 32;            // +128 B
  __bf16* ring = (__bf16*)((char*)d_ws + 1024);     // RING * 16 KB

  flags_init<<<dim3(1), dim3(64), 0, stream>>>((unsigned*)d_ws);
  rnn2_pipe_kernel<<<dim3(2), dim3(256), 0, stream>>>(x, W_ih, W_hh, b_ih, b_hh,
                                                      h0, out, prod, cons, ring);
}
